// LoRAAttnProcessor_72473278153326
// MI455X (gfx1250) — matrix operand order repaired and checked by reading, not compiled, not run on hardware
//
#include <hip/hip_runtime.h>

typedef __bf16 bf16_t;
typedef __attribute__((ext_vector_type(16))) __bf16 v16bf;
typedef __attribute__((ext_vector_type(8)))  __bf16 v8bf;
typedef __attribute__((ext_vector_type(8)))  float  v8f;

#define BB 2
#define SS 2048
#define HIDN 1024
#define HEADS 16
#define HDIM 64
#define MTOT 4096  // B*S

// ---------- scalar conversion helpers (bit tricks, proven to compile) ----------
__device__ __forceinline__ bf16_t f2bf(float f) {
  union { float f; unsigned u; } v; v.f = f;
  unsigned u = v.u + 0x7FFFu + ((v.u >> 16) & 1u);   // round-to-nearest-even
  union { unsigned short h; bf16_t b; } o; o.h = (unsigned short)(u >> 16);
  return o.b;
}
__device__ __forceinline__ float bf2f(bf16_t b) {
  union { unsigned short h; bf16_t b; } i; i.b = b;
  union { unsigned u; float f; } o; o.u = ((unsigned)i.h) << 16;
  return o.f;
}
__device__ __forceinline__ float toF(float x)  { return x; }
__device__ __forceinline__ float toF(bf16_t x) { return bf2f(x); }

// ---------- gfx1250 async memory->LDS copy (ASYNCcnt-tracked) ----------
__device__ __forceinline__ void async_b128(const bf16_t* g, bf16_t* l) {
  unsigned loff = (unsigned)(unsigned long long)
      (__attribute__((address_space(3))) const void*)l;
  asm volatile("global_load_async_to_lds_b128 %0, %1, off"
               :: "v"(loff), "v"(g) : "memory");
}
__device__ __forceinline__ void wait_async0() {
  asm volatile("s_wait_asynccnt 0x0" ::: "memory");
}
// allow 4 in-flight copies (the stage just issued for the NEXT tile)
__device__ __forceinline__ void wait_async4() {
  asm volatile("s_wait_asynccnt 0x4" ::: "memory");
}

// ---------- fragment load helpers ----------
__device__ __forceinline__ v16bf combine16(v8bf lo, v8bf hi) {
  return __builtin_shufflevector(lo, hi, 0,1,2,3,4,5,6,7,8,9,10,11,12,13,14,15);
}
// A-fragment: 16 contiguous k-elements at p
__device__ __forceinline__ v16bf ldfragA(const bf16_t* p) {
  v8bf lo = *(const v8bf*)p;
  v8bf hi = *(const v8bf*)(p + 8);
  return combine16(lo, hi);
}
// B-fragment: 16 contiguous k-elements at p
__device__ __forceinline__ v16bf ldfragB(const bf16_t* p) {
  v8bf lo = *(const v8bf*)p;
  v8bf hi = *(const v8bf*)(p + 8);
  return combine16(lo, hi);
}
__device__ __forceinline__ v8f wmma_bf16(v16bf a, v16bf b, v8f c) {
  return __builtin_amdgcn_wmma_f32_16x16x32_bf16(false, a, false, b, (short)0, c, false, false);
}

__device__ __forceinline__ void storeOut(float* p, float v)  { *p = v; }
__device__ __forceinline__ void storeOut(bf16_t* p, float v) { *p = f2bf(v); }

// ================= one-shot fp32 -> bf16 conversion (bandwidth-trivial) =================
__global__ __launch_bounds__(256) void cvt_f32_bf16_kernel(const float* __restrict__ src,
                                                           bf16_t* __restrict__ dst) {
  size_t i = ((size_t)blockIdx.x * 256 + threadIdx.x) * 8;
  float4 f0 = *(const float4*)(src + i);
  float4 f1 = *(const float4*)(src + i + 4);
  v8bf o;
  o[0]=f2bf(f0.x); o[1]=f2bf(f0.y); o[2]=f2bf(f0.z); o[3]=f2bf(f0.w);
  o[4]=f2bf(f1.x); o[5]=f2bf(f1.y); o[6]=f2bf(f1.z); o[7]=f2bf(f1.w);
  *(v8bf*)(dst + i) = o;
}

// ================= rank-4 LoRA down: t[m,r] = sum_k x[m,k]*down[r,k] =================
template<typename T>
__global__ __launch_bounds__(256) void lora_down_kernel(const T* __restrict__ x,
                                                        const float* __restrict__ down,
                                                        float* __restrict__ t) {
  const int lane = threadIdx.x & 31, wave = threadIdx.x >> 5;
  const int m = blockIdx.x * 8 + wave;
  const T* xr = x + (size_t)m * HIDN;
  float a0 = 0.f, a1 = 0.f, a2 = 0.f, a3 = 0.f;
  for (int k = lane; k < HIDN; k += 32) {
    float xv = toF(xr[k]);
    a0 += xv * down[k];
    a1 += xv * down[HIDN + k];
    a2 += xv * down[2 * HIDN + k];
    a3 += xv * down[3 * HIDN + k];
  }
  #pragma unroll
  for (int off = 16; off >= 1; off >>= 1) {
    a0 += __shfl_down(a0, off, 32);
    a1 += __shfl_down(a1, off, 32);
    a2 += __shfl_down(a2, off, 32);
    a3 += __shfl_down(a3, off, 32);
  }
  if (lane == 0) {
    float4* tp = (float4*)&t[m * 4];
    *tp = make_float4(a0, a1, a2, a3);
  }
}

// ====== GEMM + LoRA epilogue: out[m,n] = sum_k A[m,k]*W[n,k] + t[m,:]·up[n,:] (+bias) ======
// A and W pre-converted bf16. 128x128 WG tile, 8 waves as 4(M) x 2(N), wave = 32x64 (2x4 wmma).
// Software-pipelined: async copies for tile k+1 are in flight while tile k runs on WMMA.
template<typename TO>
__global__ __launch_bounds__(256) void gemm_lora_kernel(const bf16_t* __restrict__ A,
                                                        const bf16_t* __restrict__ W,
                                                        const float* __restrict__ t,
                                                        const float* __restrict__ up,
                                                        const float* __restrict__ bias,
                                                        TO* __restrict__ out) {
  __shared__ __align__(16) bf16_t As[2][128][40];
  __shared__ __align__(16) bf16_t Bs[2][128][40];
  const int tid = threadIdx.x, lane = tid & 31, wave = tid >> 5;
  const int waveM = wave >> 1, waveN = wave & 1;
  const int m0 = blockIdx.y * 128, n0 = blockIdx.x * 128;
  const int l16 = lane & 15, lh = lane >> 4;
  const int srow = tid >> 1, scol = (tid & 1) * 16;

  const bf16_t* aSrc = A + (size_t)(m0 + srow) * HIDN + scol;
  const bf16_t* wSrc = W + (size_t)(n0 + srow) * HIDN + scol;

  const v8f zero = {0.f,0.f,0.f,0.f,0.f,0.f,0.f,0.f};
  v8f acc[2][4];
  #pragma unroll
  for (int mt = 0; mt < 2; ++mt)
    #pragma unroll
    for (int nt = 0; nt < 4; ++nt) acc[mt][nt] = zero;

  // prologue: stage tile 0 into buffer 0
  async_b128(aSrc,     &As[0][srow][scol]);
  async_b128(aSrc + 8, &As[0][srow][scol + 8]);
  async_b128(wSrc,     &Bs[0][srow][scol]);
  async_b128(wSrc + 8, &Bs[0][srow][scol + 8]);

  int buf = 0;
  for (int k0 = 0; k0 < HIDN; k0 += 32) {
    if (k0 + 32 < HIDN) {
      const int nb = buf ^ 1;
      async_b128(aSrc + k0 + 32, &As[nb][srow][scol]);
      async_b128(aSrc + k0 + 40, &As[nb][srow][scol + 8]);
      async_b128(wSrc + k0 + 32, &Bs[nb][srow][scol]);
      async_b128(wSrc + k0 + 40, &Bs[nb][srow][scol + 8]);
      wait_async4();     // oldest 4 (current tile) complete; next tile in flight
    } else {
      wait_async0();
    }
    __syncthreads();

    v16bf af[2], bfv[4];
    #pragma unroll
    for (int mt = 0; mt < 2; ++mt)
      af[mt] = ldfragA(&As[buf][waveM * 32 + mt * 16 + l16][lh * 16]);
    #pragma unroll
    for (int nt = 0; nt < 4; ++nt)
      bfv[nt] = ldfragB(&Bs[buf][waveN * 64 + nt * 16 + l16][lh * 16]);
    #pragma unroll
    for (int mt = 0; mt < 2; ++mt)
      #pragma unroll
      for (int nt = 0; nt < 4; ++nt)
        acc[mt][nt] = wmma_bf16(af[mt], bfv[nt], acc[mt][nt]);
    __syncthreads();
    buf ^= 1;
  }

  // epilogue: rank-4 LoRA + optional bias
  float4 u4[4]; float bv[4];
  #pragma unroll
  for (int nt = 0; nt < 4; ++nt) {
    int n = n0 + waveN * 64 + nt * 16 + l16;
    u4[nt] = *(const float4*)&up[n * 4];
    bv[nt] = bias ? bias[n] : 0.f;
  }
  #pragma unroll
  for (int mt = 0; mt < 2; ++mt)
    #pragma unroll
    for (int j = 0; j < 8; ++j) {
      int row = m0 + waveM * 32 + mt * 16 + lh * 8 + j;
      float4 t4 = *(const float4*)&t[row * 4];
      #pragma unroll
      for (int nt = 0; nt < 4; ++nt) {
        int n = n0 + waveN * 64 + nt * 16 + l16;
        float val = acc[mt][nt][j]
                  + t4.x * u4[nt].x + t4.y * u4[nt].y
                  + t4.z * u4[nt].z + t4.w * u4[nt].w + bv[nt];
        storeOut(&out[(size_t)row * HIDN + n], val);
      }
    }
}

// ====== flash attention: grid(16 qblocks, 32 bh), 8 waves x 16 query rows = 128 q/WG ======
__global__ __launch_bounds__(256) void flash_attn_kernel(const bf16_t* __restrict__ Q,
                                                         const bf16_t* __restrict__ Kg,
                                                         const bf16_t* __restrict__ V,
                                                         bf16_t* __restrict__ O) {
  __shared__ __align__(16) bf16_t Ks[64][72];       // [key][d]  (B-frag of Q·Kt)
  __shared__ __align__(16) bf16_t Vt[64][72];       // [d][key]  (B-frag of P·V)
  __shared__ __align__(16) bf16_t Ps[8][16][72];    // per-wave C->A layout bounce
  const int tid = threadIdx.x, lane = tid & 31, wave = tid >> 5;
  const int b = blockIdx.y >> 4, h = blockIdx.y & 15;
  const int q0 = blockIdx.x * 128;
  const int l16 = lane & 15, lh = lane >> 4;

  // resident Q A-fragments (d = 64 -> 2 k-steps of 32), loaded straight from global
  const size_t qrow = (size_t)(b * SS + q0 + wave * 16 + l16);
  const bf16_t* qp = Q + qrow * HIDN + h * HDIM + lh * 16;
  const v16bf qf0 = ldfragA(qp);
  const v16bf qf1 = ldfragA(qp + 32);

  const v8f zero = {0.f,0.f,0.f,0.f,0.f,0.f,0.f,0.f};
  v8f oacc[4];
  #pragma unroll
  for (int nt = 0; nt < 4; ++nt) oacc[nt] = zero;
  float rowmax[8], rowsum[8];
  #pragma unroll
  for (int j = 0; j < 8; ++j) { rowmax[j] = -1e30f; rowsum[j] = 0.f; }

  const int keyL = tid >> 2;          // 0..63
  const int dseg = (tid & 3) * 16;    // 0,16,32,48

  for (int kb = 0; kb < SS / 64; ++kb) {
    { // stage K via async copy (pure bf16 copy), V manually transposed
      const size_t grow = (size_t)(b * SS + kb * 64 + keyL);
      const bf16_t* kp = Kg + grow * HIDN + h * HDIM + dseg;
      async_b128(kp,     &Ks[keyL][dseg]);
      async_b128(kp + 8, &Ks[keyL][dseg + 8]);
      const bf16_t* vp = V + grow * HIDN + h * HDIM + dseg;
      v8bf v0 = *(const v8bf*)vp, v1 = *(const v8bf*)(vp + 8);
      #pragma unroll
      for (int e = 0; e < 8; ++e) Vt[dseg + e][keyL] = v0[e];
      #pragma unroll
      for (int e = 0; e < 8; ++e) Vt[dseg + 8 + e][keyL] = v1[e];
    }
    wait_async0();
    __syncthreads();

    // S = Q * K^T, wave tile 16 x 64
    v8f s[4];
    #pragma unroll
    for (int nt = 0; nt < 4; ++nt) {
      s[nt] = zero;
      v16bf kf0 = ldfragB(&Ks[nt * 16 + l16][lh * 16]);
      s[nt] = wmma_bf16(qf0, kf0, s[nt]);
      v16bf kf1 = ldfragB(&Ks[nt * 16 + l16][32 + lh * 16]);
      s[nt] = wmma_bf16(qf1, kf1, s[nt]);
      s[nt] = s[nt] * 0.125f;   // 1/sqrt(64)
    }

    // online softmax (row stats live in 16-lane half-wave groups)
    float nm[8];
    #pragma unroll
    for (int j = 0; j < 8; ++j) {
      float m = rowmax[j];
      #pragma unroll
      for (int nt = 0; nt < 4; ++nt) m = fmaxf(m, s[nt][j]);
      m = fmaxf(m, __shfl_xor(m, 1, 32));
      m = fmaxf(m, __shfl_xor(m, 2, 32));
      m = fmaxf(m, __shfl_xor(m, 4, 32));
      m = fmaxf(m, __shfl_xor(m, 8, 32));
      nm[j] = m;
    }
    float corr[8];
    #pragma unroll
    for (int j = 0; j < 8; ++j) { corr[j] = __expf(rowmax[j] - nm[j]); rowmax[j] = nm[j]; }
    #pragma unroll
    for (int nt = 0; nt < 4; ++nt)
      #pragma unroll
      for (int j = 0; j < 8; ++j) s[nt][j] = __expf(s[nt][j] - nm[j]);
    #pragma unroll
    for (int j = 0; j < 8; ++j) {
      float ts = s[0][j] + s[1][j] + s[2][j] + s[3][j];
      ts += __shfl_xor(ts, 1, 32);
      ts += __shfl_xor(ts, 2, 32);
      ts += __shfl_xor(ts, 4, 32);
      ts += __shfl_xor(ts, 8, 32);
      rowsum[j] = rowsum[j] * corr[j] + ts;
    }
    #pragma unroll
    for (int nt = 0; nt < 4; ++nt)
      #pragma unroll
      for (int j = 0; j < 8; ++j) oacc[nt][j] *= corr[j];

    // C-layout -> A-layout via per-wave LDS bounce
    #pragma unroll
    for (int nt = 0; nt < 4; ++nt)
      #pragma unroll
      for (int j = 0; j < 8; ++j)
        Ps[wave][lh * 8 + j][nt * 16 + l16] = f2bf(s[nt][j]);

    v16bf pf0 = ldfragA(&Ps[wave][l16][lh * 16]);
    v16bf pf1 = ldfragA(&Ps[wave][l16][32 + lh * 16]);
    #pragma unroll
    for (int nt = 0; nt < 4; ++nt) {
      v16bf vf0 = ldfragB(&Vt[nt * 16 + l16][lh * 16]);
      oacc[nt] = wmma_bf16(pf0, vf0, oacc[nt]);
      v16bf vf1 = ldfragB(&Vt[nt * 16 + l16][32 + lh * 16]);
      oacc[nt] = wmma_bf16(pf1, vf1, oacc[nt]);
    }
    __syncthreads();
  }

  #pragma unroll
  for (int j = 0; j < 8; ++j) {
    float inv = 1.0f / rowsum[j];
    int row = q0 + wave * 16 + lh * 8 + j;
    size_t base = (size_t)(b * SS + row) * HIDN + h * HDIM;
    #pragma unroll
    for (int nt = 0; nt < 4; ++nt)
      O[base + nt * 16 + l16] = f2bf(oacc[nt][j] * inv);
  }
}

extern "C" void kernel_launch(void* const* d_in, const int* in_sizes, int n_in,
                              void* d_out, int out_size, void* d_ws, size_t ws_size,
                              hipStream_t stream) {
  (void)in_sizes; (void)n_in; (void)out_size; (void)ws_size;
  const float* x      = (const float*)d_in[0];
  const float* wq     = (const float*)d_in[1];
  const float* wk     = (const float*)d_in[2];
  const float* wv     = (const float*)d_in[3];
  const float* wo     = (const float*)d_in[4];
  const float* bo     = (const float*)d_in[5];
  const float* q_down = (const float*)d_in[6];
  const float* q_up   = (const float*)d_in[7];
  const float* k_down = (const float*)d_in[8];
  const float* k_up   = (const float*)d_in[9];
  const float* v_down = (const float*)d_in[10];
  const float* v_up   = (const float*)d_in[11];
  const float* o_down = (const float*)d_in[12];
  const float* o_up   = (const float*)d_in[13];

  char* ws = (char*)d_ws;
  const size_t SZ  = (size_t)MTOT * HIDN * sizeof(bf16_t);   // 8 MB
  const size_t WSZ = (size_t)HIDN * HIDN * sizeof(bf16_t);   // 2 MB
  bf16_t* xb  = (bf16_t*)(ws);
  bf16_t* qb  = (bf16_t*)(ws + SZ);
  bf16_t* kb  = (bf16_t*)(ws + 2 * SZ);
  bf16_t* vb  = (bf16_t*)(ws + 3 * SZ);
  bf16_t* ab  = (bf16_t*)(ws + 4 * SZ);
  bf16_t* wqb = (bf16_t*)(ws + 5 * SZ);
  bf16_t* wkb = (bf16_t*)(ws + 5 * SZ + WSZ);
  bf16_t* wvb = (bf16_t*)(ws + 5 * SZ + 2 * WSZ);
  bf16_t* wob = (bf16_t*)(ws + 5 * SZ + 3 * WSZ);
  float*  tq  = (float*)(ws + 5 * SZ + 4 * WSZ);
  float*  tk  = tq + MTOT * 4;
  float*  tv  = tk + MTOT * 4;
  float*  to  = tv + MTOT * 4;

  dim3 blk(256);
  dim3 gg(HIDN / 128, MTOT / 128);   // (8, 32)

  // one-shot bf16 conversions (x: 4M elems, weights: 1M elems each)
  cvt_f32_bf16_kernel<<<MTOT * HIDN / 2048, blk, 0, stream>>>(x,  xb);
  cvt_f32_bf16_kernel<<<HIDN * HIDN / 2048, blk, 0, stream>>>(wq, wqb);
  cvt_f32_bf16_kernel<<<HIDN * HIDN / 2048, blk, 0, stream>>>(wk, wkb);
  cvt_f32_bf16_kernel<<<HIDN * HIDN / 2048, blk, 0, stream>>>(wv, wvb);
  cvt_f32_bf16_kernel<<<HIDN * HIDN / 2048, blk, 0, stream>>>(wo, wob);

  lora_down_kernel<float><<<MTOT / 8, blk, 0, stream>>>(x, q_down, tq);
  lora_down_kernel<float><<<MTOT / 8, blk, 0, stream>>>(x, k_down, tk);
  lora_down_kernel<float><<<MTOT / 8, blk, 0, stream>>>(x, v_down, tv);

  gemm_lora_kernel<bf16_t><<<gg, blk, 0, stream>>>(xb, wqb, tq, q_up, nullptr, qb);
  gemm_lora_kernel<bf16_t><<<gg, blk, 0, stream>>>(xb, wkb, tk, k_up, nullptr, kb);
  gemm_lora_kernel<bf16_t><<<gg, blk, 0, stream>>>(xb, wvb, tv, v_up, nullptr, vb);

  flash_attn_kernel<<<dim3(SS / 128, BB * HEADS), blk, 0, stream>>>(qb, kb, vb, ab);

  lora_down_kernel<bf16_t><<<MTOT / 8, blk, 0, stream>>>(ab, o_down, to);
  gemm_lora_kernel<float><<<gg, blk, 0, stream>>>(ab, wob, to, o_up, bo, (float*)d_out);
}
